// SoftPoolingGcnEncoder_45818711113830
// MI455X (gfx1250) — compile-verified, matrix operand order checked by reading, not run on hardware
//
#include <hip/hip_runtime.h>
#include <hip/hip_bf16.h>

#define BB 4
#define NN 512
#define CC 64
#define HH 128
#define T_THRESH 10
#define K_DIL 2

typedef float v2f __attribute__((ext_vector_type(2)));
typedef float v8f __attribute__((ext_vector_type(8)));
typedef _Float16 v16h __attribute__((ext_vector_type(16)));

static __device__ __forceinline__ unsigned ballot32(bool p) {
#if __has_builtin(__builtin_amdgcn_ballot_w32)
  return __builtin_amdgcn_ballot_w32(p);
#else
  return (unsigned)__ballot(p);
#endif
}

// ---------------------------------------------------------------------------
// Kernel 1: per-row dilation decisions -> "removed" bitmask (B*N x 16 words)
// One wave32 per adjacency row.
// ---------------------------------------------------------------------------
__global__ void k_dilate(const float* __restrict__ adj,
                         const unsigned char* __restrict__ mask,
                         unsigned* __restrict__ R) {
  int wid  = (blockIdx.x * blockDim.x + threadIdx.x) >> 5;   // row id in [0, B*N)
  int lane = threadIdx.x & 31;
  if (wid >= BB * NN) return;
  int i = wid & (NN - 1);
  const float* arow = adj + (size_t)wid * NN;

  // pass 1: neighbor count m (self excluded; adj_ has forced diag=1 but nb excludes it)
  int m = 0;
  for (int ch = 0; ch < NN / 32; ++ch) {
    int j = ch * 32 + lane;
    bool nb = (j != i) && (arow[j] > 0.0f);
    m += __popc(ballot32(nb));
  }
  int skip = (m > T_THRESH) ? ((m + K_DIL - 1) / K_DIL) : 1;
  bool do_dil = (skip > 1) && (mask[wid] != 0);

  // pass 2: rank among neighbors -> removed bit
  int base = 0;
  for (int ch = 0; ch < NN / 32; ++ch) {
    int j = ch * 32 + lane;
    bool nb = (j != i) && (arow[j] > 0.0f);
    unsigned msk = ballot32(nb);
    int rank = base + __popc(msk & ((1u << lane) - 1u));      // 0-based among neighbors
    bool removed = nb && do_dil && (((rank + 1) % skip) == 0);
    unsigned rw = ballot32(removed);
    if (lane == 0) R[(size_t)wid * 16 + ch] = rw;
    base += __popc(msk);
  }
}

// ---------------------------------------------------------------------------
// Kernel 2: valid = (adj_ > 0) && !(removed || removed^T), packed bitmask.
// ---------------------------------------------------------------------------
__global__ void k_valid(const float* __restrict__ adj,
                        const unsigned* __restrict__ R,
                        unsigned* __restrict__ V) {
  int j = blockIdx.x * blockDim.x + threadIdx.x;   // column
  int i = blockIdx.y;                              // row
  int b = blockIdx.z;
  bool a  = (j == i) || (adj[((size_t)b * NN + i) * NN + j] > 0.0f);
  bool r1 = (R[((size_t)b * NN + i) * 16 + (j >> 5)] >> (j & 31)) & 1u;
  bool r2 = (R[((size_t)b * NN + j) * 16 + (i >> 5)] >> (i & 31)) & 1u;
  bool v = a && !r1 && !r2;
  unsigned w = ballot32(v);
  if ((threadIdx.x & 31) == 0) V[((size_t)b * NN + i) * 16 + (j >> 5)] = w;
}

// ---------------------------------------------------------------------------
// Kernel 3: weighted quantile aggregation + residual.  One 64-thread block
// per (b,n); thread t owns channel t.  Neighbor values staged in dynamic LDS
// as vals[a*64 + c] (bank == channel -> conflict-free).  Order statistics by
// rank counting (tie-equivalent to sort+gather).
// ---------------------------------------------------------------------------
__global__ void k_quantile(const float* __restrict__ x,
                           const unsigned* __restrict__ V,
                           float* __restrict__ outBuf) {
  extern __shared__ float vals[];                  // up to 512*64 floats
  __shared__ unsigned short s_list[NN];
  __shared__ int s_cnt;

  int bn = blockIdx.x;                             // (b,n) flat
  int b  = bn >> 9;
  int t  = threadIdx.x;                            // channel 0..63

  if (t == 0) s_cnt = 0;
  __syncthreads();

  const unsigned* vrow = V + (size_t)bn * 16;
  for (int ch = 0; ch < NN / CC; ++ch) {           // 8 strided sweeps
    int j = ch * CC + t;
    if ((vrow[j >> 5] >> (j & 31)) & 1u) {
      int p = atomicAdd(&s_cnt, 1);
      s_list[p] = (unsigned short)j;
    }
  }
  __syncthreads();
  int m = s_cnt;

  const float* xb = x + (size_t)b * NN * CC;
  for (int a = 0; a < m; ++a)                      // coalesced row loads
    vals[a * CC + t] = xb[(size_t)s_list[a] * CC + t];
  __syncthreads();

  float agg = 0.0f;
  if (m > 0) {
    const float taus[3] = {0.25f, 0.5f, 0.75f};
    const float wts[3]  = {0.25f, 0.5f, 0.25f};
    int lo[3], hi[3];
    float fr[3];
    float mf = (float)m;
#pragma unroll
    for (int q = 0; q < 3; ++q) {
      float pos = taus[q] * (mf - 1.0f);
      if (pos < 0.0f) pos = 0.0f;
      float fl = floorf(pos);
      lo[q] = (int)fl;
      hi[q] = (int)ceilf(pos);
      fr[q] = pos - fl;
    }
    float qlo[3] = {0.f, 0.f, 0.f}, qhi[3] = {0.f, 0.f, 0.f};
    for (int a = 0; a < m; ++a) {
      float v = vals[a * CC + t];
      int less = 0, eq = 0;
      for (int k = 0; k < m; ++k) {
        float w = vals[k * CC + t];
        less += (w < v);
        eq   += (w == v);
      }
#pragma unroll
      for (int q = 0; q < 3; ++q) {
        if (less <= lo[q] && lo[q] < less + eq) qlo[q] = v;
        if (less <= hi[q] && hi[q] < less + eq) qhi[q] = v;
      }
    }
#pragma unroll
    for (int q = 0; q < 3; ++q)
      agg += wts[q] * (qlo[q] * (1.0f - fr[q]) + qhi[q] * fr[q]);
  }
  outBuf[(size_t)bn * CC + t] = xb[((size_t)(bn & (NN - 1))) * CC + t] + agg;
}

// ---------------------------------------------------------------------------
// Kernel 4: fused GIN MLP with WMMA.  One wave32 per 16-row tile.
// Prefer full-precision V_WMMA_F32_16X16X4_F32 (matches f32 reference; the
// GEMMs are ~67 MFLOP total, nowhere near any roofline, so precision is free).
// ---------------------------------------------------------------------------
#if __has_builtin(__builtin_amdgcn_wmma_f32_16x16x4_f32)
#define KSTEP 4
static __device__ __forceinline__ v8f mma_step(const float* __restrict__ Arow,
                                               const float* __restrict__ Bm,
                                               int ldb, int k0, int col,
                                               int kgrp, v8f acc) {
  v2f a, bb;
  a.x = Arow[k0 + kgrp * 2];
  a.y = Arow[k0 + kgrp * 2 + 1];
  bb.x = Bm[(k0 + kgrp * 2) * ldb + col];
  bb.y = Bm[(k0 + kgrp * 2 + 1) * ldb + col];
  return __builtin_amdgcn_wmma_f32_16x16x4_f32(false, a, false, bb,
                                               (short)0, acc, false, false);
}
#else
#define KSTEP 32
static __device__ __forceinline__ v8f mma_step(const float* __restrict__ Arow,
                                               const float* __restrict__ Bm,
                                               int ldb, int k0, int col,
                                               int kgrp, v8f acc) {
  v16h a, bb;
#pragma unroll
  for (int v = 0; v < 8; ++v) {
    int kb = (v < 4) ? (v * 2 + kgrp * 8) : (16 + (v - 4) * 2 + kgrp * 8);
    a[2 * v]      = (_Float16)Arow[k0 + kb];
    a[2 * v + 1]  = (_Float16)Arow[k0 + kb + 1];
    bb[2 * v]     = (_Float16)Bm[(k0 + kb) * ldb + col];
    bb[2 * v + 1] = (_Float16)Bm[(k0 + kb + 1) * ldb + col];
  }
  return __builtin_amdgcn_wmma_f32_16x16x32_f16(false, a, false, bb,
                                                (short)0, acc, false, false);
}
#endif

__global__ void __launch_bounds__(128) k_mlp(const float* __restrict__ outBuf,
                      const float* __restrict__ W1, const float* __restrict__ b1,
                      const float* __restrict__ W2, const float* __restrict__ b2,
                      const unsigned char* __restrict__ mask,
                      float* __restrict__ y) {
  __shared__ float sh[4][16 * HH];                 // per-wave h tile (16x128)

  int lane = threadIdx.x & 31;
  int wv   = threadIdx.x >> 5;
  int tile = blockIdx.x * 4 + wv;                  // 16-row tile id
  int rowBase = tile * 16;
  int mrow = lane & 15;
  int kgrp = lane >> 4;
  int col  = lane & 15;

  __builtin_prefetch(W1, 0, 3);
  __builtin_prefetch(W2, 0, 3);

  const float* Arow = outBuf + (size_t)(rowBase + mrow) * CC;
  float* hs = &sh[wv][0];

  // GEMM1: (16 x 64) @ (64 x 128) -> relu -> LDS
  for (int nt = 0; nt < HH / 16; ++nt) {
    v8f acc = {};
    for (int k0 = 0; k0 < CC; k0 += KSTEP)
      acc = mma_step(Arow, W1, HH, k0, nt * 16 + col, kgrp, acc);
    float bias = b1[nt * 16 + col];
#pragma unroll
    for (int r = 0; r < 8; ++r) {
      int M = r + kgrp * 8;                        // C/D layout: lanes16-31 hold M+8
      float hv = acc[r] + bias;
      hs[M * HH + nt * 16 + col] = fmaxf(hv, 0.0f);
    }
  }
  __syncthreads();

  // GEMM2: (16 x 128) @ (128 x 64) + bias, masked store
  const float* Hrow = hs + mrow * HH;
  for (int ct = 0; ct < CC / 16; ++ct) {
    v8f acc = {};
    for (int k0 = 0; k0 < HH; k0 += KSTEP)
      acc = mma_step(Hrow, W2, CC, k0, ct * 16 + col, kgrp, acc);
    float bias = b2[ct * 16 + col];
#pragma unroll
    for (int r = 0; r < 8; ++r) {
      int M = r + kgrp * 8;
      int row = rowBase + M;                       // flat (b,n)
      float mk = mask[row] ? 1.0f : 0.0f;
      y[(size_t)row * CC + ct * 16 + col] = (acc[r] + bias) * mk;
    }
  }
}

// ---------------------------------------------------------------------------
extern "C" void kernel_launch(void* const* d_in, const int* in_sizes, int n_in,
                              void* d_out, int out_size, void* d_ws, size_t ws_size,
                              hipStream_t stream) {
  (void)in_sizes; (void)n_in; (void)out_size; (void)ws_size;
  const float*         x    = (const float*)d_in[0];          // (4,512,64)
  const float*         adj  = (const float*)d_in[1];          // (4,512,512)
  const unsigned char* mask = (const unsigned char*)d_in[2];  // (4,512) bool
  const float*         W1   = (const float*)d_in[3];          // (64,128)
  const float*         b1   = (const float*)d_in[4];          // (128)
  const float*         W2   = (const float*)d_in[5];          // (128,64)
  const float*         b2   = (const float*)d_in[6];          // (64)
  float*               y    = (float*)d_out;                  // (4,512,64)

  const int ROWS = BB * NN;                                   // 2048
  unsigned* R      = (unsigned*)d_ws;                         // 2048*16 words
  unsigned* V      = R + (size_t)ROWS * 16;                   // 2048*16 words
  float*    outBuf = (float*)(V + (size_t)ROWS * 16);         // 2048*64 floats

  // 1) dilation decisions (one wave per row; 4 waves/block)
  k_dilate<<<dim3(ROWS / 4), dim3(128), 0, stream>>>(adj, mask, R);

  // 2) symmetrize -> valid bitmask
  k_valid<<<dim3(NN / 256, NN, BB), dim3(256), 0, stream>>>(adj, R, V);

  // 3) quantile aggregation + residual (dynamic LDS: 512*64 floats)
  size_t lds = (size_t)NN * CC * sizeof(float);
  k_quantile<<<dim3(ROWS), dim3(CC), lds, stream>>>(x, V, outBuf);

  // 4) fused WMMA MLP (32 blocks x 4 waves x 16 rows = 2048 rows)
  k_mlp<<<dim3(ROWS / 64), dim3(128), 0, stream>>>(outBuf, W1, b1, W2, b2, mask, y);
}
